// CamadaEquivariante_52699248722544
// MI455X (gfx1250) — matrix operand order repaired and checked by reading, not compiled, not run on hardware
//
#include <hip/hip_runtime.h>
#include <math.h>

typedef _Float16 f16;
typedef _Float16 v16h __attribute__((ext_vector_type(16)));
typedef _Float16 v8h  __attribute__((ext_vector_type(8)));
typedef float    v8f  __attribute__((ext_vector_type(8)));

#define WV  8   // waves per block
#define TPW 2   // 16-row tiles per wave

// ---------------------------------------------------------------- helpers ---

__device__ __forceinline__ v8f wmma_f16(v16h a, v16h b, v8f c) {
  // v_wmma_f32_16x16x32_f16: (neg_a, A, neg_b, B, c_mod, C, reuse_a, reuse_b)
  return __builtin_amdgcn_wmma_f32_16x16x32_f16(false, a, false, b, (short)0, c,
                                                false, false);
}

__device__ __forceinline__ void lds_fence() {
  // CDNA5 split counters: wait for this wave's outstanding LDS ops
  asm volatile("s_wait_dscnt 0" ::: "memory");
}

// CDNA5 hardware tanh (TRANS32 pipe, co-executes with matrix ops)
__device__ __forceinline__ float fast_tanh(float v) {
#if __has_builtin(__builtin_amdgcn_tanhf)
  return __builtin_amdgcn_tanhf(v);
#elif defined(__gfx1250__)
  float r;
  asm("v_tanh_f32 %0, %1" : "=v"(r) : "v"(v));
  return r;
#else
  return tanhf(v);
#endif
}

__device__ __forceinline__ v8f vzero8() {
  v8f z;
#pragma unroll
  for (int i = 0; i < 8; ++i) z[i] = 0.0f;
  return z;
}

// A fragment (16x32 f16, M x K) from row-major LDS tile [16][ldk].
// Per ISA 7.12.2: lane m = lane&15; lanes<16 hold K {0..7,16..23}, lanes>=16
// hold K {8..15,24..31} within the 32-wide K chunk.
__device__ __forceinline__ v16h load_a(const f16* Xs, int ldk, int lane, int kbase) {
  const f16* p = Xs + (lane & 15) * ldk + kbase + ((lane >> 4) << 3);
  v8h lo = *(const v8h*)p;
  v8h hi = *(const v8h*)(p + 16);
  v16h a;
#pragma unroll
  for (int t = 0; t < 8; ++t) { a[t] = lo[t]; a[8 + t] = hi[t]; }
  return a;
}

// B fragment from pre-swizzled LDS weights: one contiguous 32B load per lane.
__device__ __forceinline__ v16h load_b(const f16* Wswz, int kc, int nt, int lane) {
  return *(const v16h*)(Wswz + (((kc * 4 + nt) * 32 + lane) << 4));
}

// Pre-swizzle a global f32 weight W[Kreal][64] into B-fragment layout, f16,
// zero-padding K up to nkc*32. Element t of lane in chunk (kc,nt) is
// B[kc*32 + (t&7) + 8*(lane>>4) + 16*(t>>3)][nt*16 + (lane&15)].
__device__ void fill_wswz(f16* dst, const float* __restrict__ W, int Kreal,
                          int nkc, int tid, int nthreads) {
  int total = nkc * 2048;
  for (int idx = tid; idx < total; idx += nthreads) {
    int t    = idx & 15;
    int lane = (idx >> 4) & 31;
    int nt   = (idx >> 9) & 3;
    int kc   = idx >> 11;
    int k = kc * 32 + (t & 7) + ((lane >> 4) << 3) + ((t >> 3) << 4);
    int n = nt * 16 + (lane & 15);
    dst[idx] = (k < Kreal) ? (f16)W[k * 64 + n] : (f16)0.0f;
  }
}

// ------------------------------------------------------------- init kernel --

__global__ void zero_kernel(float* __restrict__ p, long n) {
  long i = (long)blockIdx.x * blockDim.x + threadIdx.x;
  if (i < n) p[i] = 0.0f;
}

// ------------------------------------------------------------- edge kernel --
// Per 16-edge tile: X = [h_i(64) | h_j(64) | r^2(1) | attr(8) | 0 pad] (160)
// m_ij = tanh(tanh(X@We1+be1)@We2+be2)      -> atomicAdd into m_i[N,64]
// phi  = tanh(tanh(m_ij@Wx1+bx1)@wx2+bx2)   -> atomicAdd diff*phi, cnt

__global__ __launch_bounds__(256) void edge_kernel(
    const float* __restrict__ h, const float* __restrict__ x,
    const float* __restrict__ eattr, const int* __restrict__ arestas,
    const float* __restrict__ we_w1, const float* __restrict__ we_b1,
    const float* __restrict__ we_w2, const float* __restrict__ we_b2,
    const float* __restrict__ wx_w1, const float* __restrict__ wx_b1,
    const float* __restrict__ wx_w2, const float* __restrict__ wx_b2,
    float* __restrict__ m_i, float* __restrict__ aggsum, float* __restrict__ cnt,
    int N, int E) {
  extern __shared__ char smem[];
  f16*   sWe1 = (f16*)smem;            // 5*2048 halfs (K padded 137->160)
  f16*   sWe2 = sWe1 + 10240;          // 2*2048
  f16*   sWx1 = sWe2 + 4096;           // 2*2048
  float* sF   = (float*)(sWx1 + 4096);
  float* sBe1 = sF;                    // 64
  float* sBe2 = sF + 64;               // 64
  float* sBx1 = sF + 128;              // 64
  float* sWx2 = sF + 192;              // 64
  f16*   sX   = (f16*)(sF + 256);      // WV * 16*160
  f16*   sAct = sX + WV * 2560;        // WV * 16*64
  float* sT2  = (float*)(sAct + WV * 1024); // WV * 16*64 f32
  int*   sNode = (int*)(sT2 + WV * 1024);   // WV * 16

  const int tid  = threadIdx.x;
  const int lane = tid & 31;
  const int wave = tid >> 5;

  fill_wswz(sWe1, we_w1, 137, 5, tid, 256);
  fill_wswz(sWe2, we_w2,  64, 2, tid, 256);
  fill_wswz(sWx1, wx_w1,  64, 2, tid, 256);
  if (tid < 64) {
    sBe1[tid] = we_b1[tid];
    sBe2[tid] = we_b2[tid];
    sBx1[tid] = wx_b1[tid];
    sWx2[tid] = wx_w2[tid];
  }
  __syncthreads();
  const float bx2 = wx_b2[0];

  const int* __restrict__ linhas = arestas;
  const int* __restrict__ colsI  = arestas + E;

  f16*   myX   = sX   + wave * 2560;
  f16*   myAct = sAct + wave * 1024;
  float* myT2  = sT2  + wave * 1024;
  int*   myNode = sNode + wave * 16;

  const int row  = lane & 15;
  const int part = lane >> 4;
  const int ncol = lane & 15;
  const int Moff = part << 3;

  for (int tt = 0; tt < TPW; ++tt) {
    const long tile  = (long)blockIdx.x * (WV * TPW) + wave * TPW + tt;
    const long ebase = tile * 16;
    const long e     = ebase + row;
    const bool valid = (e < (long)E);
    const long ec    = valid ? e : 0;

    const int i = linhas[ec];
    const int j = colsI[ec];
    if (part == 0) myNode[row] = i;

    const float dx = x[i * 3 + 0] - x[j * 3 + 0];
    const float dy = x[i * 3 + 1] - x[j * 3 + 1];
    const float dz = x[i * 3 + 2] - x[j * 3 + 2];
    const float rad = dx * dx + dy * dy + dz * dz;

    // stage input tile: part0 -> h_i (cols 0..63), part1 -> h_j + tail
    {
      const float* hp = (part == 0) ? (h + (long)i * 64) : (h + (long)j * 64);
      f16* dst = myX + row * 160 + part * 64;
#pragma unroll 16
      for (int c = 0; c < 64; ++c) dst[c] = (f16)hp[c];
    }
    if (part == 1) {
      f16* dst = myX + row * 160;
      dst[128] = (f16)rad;
#pragma unroll
      for (int k = 0; k < 8; ++k) dst[129 + k] = (f16)eattr[ec * 8 + k];
#pragma unroll
      for (int k = 137; k < 160; ++k) dst[k] = (f16)0.0f;
    }
    lds_fence();

    // ---- layer 1: [16x160] @ [160x64] ----
    v8f acc[4];
#pragma unroll
    for (int nt = 0; nt < 4; ++nt) acc[nt] = vzero8();
#pragma unroll
    for (int kc = 0; kc < 5; ++kc) {
      v16h a = load_a(myX, 160, lane, kc * 32);
#pragma unroll
      for (int nt = 0; nt < 4; ++nt)
        acc[nt] = wmma_f16(a, load_b(sWe1, kc, nt, lane), acc[nt]);
    }
#pragma unroll
    for (int nt = 0; nt < 4; ++nt) {
      const float bias = sBe1[nt * 16 + ncol];
#pragma unroll
      for (int r = 0; r < 8; ++r) {
        float v = fast_tanh(acc[nt][r] + bias);
        myAct[(r + Moff) * 64 + nt * 16 + ncol] = (f16)v;
      }
    }
    lds_fence();

    // ---- layer 2: [16x64] @ [64x64] -> m_ij ----
#pragma unroll
    for (int nt = 0; nt < 4; ++nt) acc[nt] = vzero8();
#pragma unroll
    for (int kc = 0; kc < 2; ++kc) {
      v16h a = load_a(myAct, 64, lane, kc * 32);
#pragma unroll
      for (int nt = 0; nt < 4; ++nt)
        acc[nt] = wmma_f16(a, load_b(sWe2, kc, nt, lane), acc[nt]);
    }
#pragma unroll
    for (int nt = 0; nt < 4; ++nt) {
      const float bias = sBe2[nt * 16 + ncol];
#pragma unroll
      for (int r = 0; r < 8; ++r) {
        const int   M = r + Moff;
        const float v = fast_tanh(acc[nt][r] + bias);
        if (ebase + M < (long)E)
          atomicAdd(&m_i[(long)myNode[M] * 64 + nt * 16 + ncol], v);
        myAct[M * 64 + nt * 16 + ncol] = (f16)v;  // reuse as phi_x input
      }
    }
    lds_fence();

    // ---- phi_x layer 1: [16x64] @ [64x64] ----
#pragma unroll
    for (int nt = 0; nt < 4; ++nt) acc[nt] = vzero8();
#pragma unroll
    for (int kc = 0; kc < 2; ++kc) {
      v16h a = load_a(myAct, 64, lane, kc * 32);
#pragma unroll
      for (int nt = 0; nt < 4; ++nt)
        acc[nt] = wmma_f16(a, load_b(sWx1, kc, nt, lane), acc[nt]);
    }
#pragma unroll
    for (int nt = 0; nt < 4; ++nt) {
      const float bias = sBx1[nt * 16 + ncol];
#pragma unroll
      for (int r = 0; r < 8; ++r)
        myT2[(r + Moff) * 64 + nt * 16 + ncol] = fast_tanh(acc[nt][r] + bias);
    }
    lds_fence();

    // ---- phi_x layer 2 (64 -> 1) + coordinate aggregation ----
    if (part == 0) {
      float s = 0.0f;
      for (int c = 0; c < 64; ++c) s += myT2[row * 64 + c] * sWx2[c];
      const float phi = fast_tanh(s + bx2);
      if (valid) {
        atomicAdd(&aggsum[(long)i * 3 + 0], dx * phi);
        atomicAdd(&aggsum[(long)i * 3 + 1], dy * phi);
        atomicAdd(&aggsum[(long)i * 3 + 2], dz * phi);
        atomicAdd(&cnt[i], 1.0f);
      }
    }
  }
}

// ------------------------------------------------------------- node kernel --
// phi_v = tanh(h@Wv1+bv1)@wv2 + bv2
// vel'  = vel*phi_v + aggsum/max(cnt,1);  x' = x + vel'
// h'    = tanh([h|m_i]@Wh1+bh1)@Wh2 + bh2

__global__ __launch_bounds__(256) void node_kernel(
    const float* __restrict__ h, const float* __restrict__ x,
    const float* __restrict__ vel,
    const float* __restrict__ wh_w1, const float* __restrict__ wh_b1,
    const float* __restrict__ wh_w2, const float* __restrict__ wh_b2,
    const float* __restrict__ wv_w1, const float* __restrict__ wv_b1,
    const float* __restrict__ wv_w2, const float* __restrict__ wv_b2,
    const float* __restrict__ m_i, const float* __restrict__ aggsum,
    const float* __restrict__ cnt, float* __restrict__ out, int N) {
  extern __shared__ char smem[];
  f16*   sWh1 = (f16*)smem;            // 4*2048 (K = 128)
  f16*   sWh2 = sWh1 + 8192;           // 2*2048
  f16*   sWv1 = sWh2 + 4096;           // 2*2048
  float* sF   = (float*)(sWv1 + 4096);
  float* sBh1 = sF;
  float* sBh2 = sF + 64;
  float* sBv1 = sF + 128;
  float* sWv2 = sF + 192;
  f16*   sX   = (f16*)(sF + 256);      // WV * 16*128
  f16*   sAct = sX + WV * 2048;        // WV * 16*64
  float* sT2  = (float*)(sAct + WV * 1024); // WV * 16*64 f32

  const int tid  = threadIdx.x;
  const int lane = tid & 31;
  const int wave = tid >> 5;

  fill_wswz(sWh1, wh_w1, 128, 4, tid, 256);
  fill_wswz(sWh2, wh_w2,  64, 2, tid, 256);
  fill_wswz(sWv1, wv_w1,  64, 2, tid, 256);
  if (tid < 64) {
    sBh1[tid] = wh_b1[tid];
    sBh2[tid] = wh_b2[tid];
    sBv1[tid] = wv_b1[tid];
    sWv2[tid] = wv_w2[tid];
  }
  __syncthreads();
  const float bv2 = wv_b2[0];

  float* __restrict__ out_h = out;
  float* __restrict__ out_x = out + (long)N * 64;
  float* __restrict__ out_v = out + (long)N * 67;

  f16*   myX   = sX   + wave * 2048;
  f16*   myAct = sAct + wave * 1024;
  float* myT2  = sT2  + wave * 1024;

  const int row  = lane & 15;
  const int part = lane >> 4;
  const int ncol = lane & 15;
  const int Moff = part << 3;

  for (int tt = 0; tt < TPW; ++tt) {
    const long tile  = (long)blockIdx.x * (WV * TPW) + wave * TPW + tt;
    const long nbase = tile * 16;
    const long n     = nbase + row;
    const bool valid = (n < (long)N);
    const long nc    = valid ? n : 0;

    // stage [h | m_i] tile (16 x 128)
    {
      const float* src = (part == 0) ? (h + nc * 64) : (m_i + nc * 64);
      f16* dst = myX + row * 128 + part * 64;
#pragma unroll 16
      for (int c = 0; c < 64; ++c) dst[c] = (f16)src[c];
    }
    lds_fence();

    // ---- phi_v layer 1 over h (K=64 from the 128-wide tile) ----
    v8f acc[4];
#pragma unroll
    for (int nt = 0; nt < 4; ++nt) acc[nt] = vzero8();
#pragma unroll
    for (int kc = 0; kc < 2; ++kc) {
      v16h a = load_a(myX, 128, lane, kc * 32);
#pragma unroll
      for (int nt = 0; nt < 4; ++nt)
        acc[nt] = wmma_f16(a, load_b(sWv1, kc, nt, lane), acc[nt]);
    }
#pragma unroll
    for (int nt = 0; nt < 4; ++nt) {
      const float bias = sBv1[nt * 16 + ncol];
#pragma unroll
      for (int r = 0; r < 8; ++r)
        myT2[(r + Moff) * 64 + nt * 16 + ncol] = fast_tanh(acc[nt][r] + bias);
    }
    lds_fence();

    // ---- phi_v layer 2 + velocity/coordinate update ----
    if (part == 0) {
      float s = 0.0f;
      for (int c = 0; c < 64; ++c) s += myT2[row * 64 + c] * sWv2[c];
      const float phiv = s + bv2;  // no outer tanh
      if (valid) {
        const float dnm = fmaxf(cnt[n], 1.0f);
#pragma unroll
        for (int k = 0; k < 3; ++k) {
          const float mean = aggsum[n * 3 + k] / dnm;
          const float vn   = vel[n * 3 + k] * phiv + mean;
          out_v[n * 3 + k] = vn;
          out_x[n * 3 + k] = x[n * 3 + k] + vn;
        }
      }
    }

    // ---- h layer 1: [16x128] @ [128x64] ----
#pragma unroll
    for (int nt = 0; nt < 4; ++nt) acc[nt] = vzero8();
#pragma unroll
    for (int kc = 0; kc < 4; ++kc) {
      v16h a = load_a(myX, 128, lane, kc * 32);
#pragma unroll
      for (int nt = 0; nt < 4; ++nt)
        acc[nt] = wmma_f16(a, load_b(sWh1, kc, nt, lane), acc[nt]);
    }
#pragma unroll
    for (int nt = 0; nt < 4; ++nt) {
      const float bias = sBh1[nt * 16 + ncol];
#pragma unroll
      for (int r = 0; r < 8; ++r)
        myAct[(r + Moff) * 64 + nt * 16 + ncol] = (f16)fast_tanh(acc[nt][r] + bias);
    }
    lds_fence();

    // ---- h layer 2: [16x64] @ [64x64] (no outer tanh) ----
#pragma unroll
    for (int nt = 0; nt < 4; ++nt) acc[nt] = vzero8();
#pragma unroll
    for (int kc = 0; kc < 2; ++kc) {
      v16h a = load_a(myAct, 64, lane, kc * 32);
#pragma unroll
      for (int nt = 0; nt < 4; ++nt)
        acc[nt] = wmma_f16(a, load_b(sWh2, kc, nt, lane), acc[nt]);
    }
#pragma unroll
    for (int nt = 0; nt < 4; ++nt) {
      const float bias = sBh2[nt * 16 + ncol];
#pragma unroll
      for (int r = 0; r < 8; ++r) {
        const int M = r + Moff;
        if (nbase + M < (long)N)
          out_h[(nbase + M) * 64 + nt * 16 + ncol] = acc[nt][r] + bias;
      }
    }
  }
}

// ---------------------------------------------------------------- launcher --

extern "C" void kernel_launch(void* const* d_in, const int* in_sizes, int n_in,
                              void* d_out, int out_size, void* d_ws, size_t ws_size,
                              hipStream_t stream) {
  const float* h     = (const float*)d_in[0];
  const float* x     = (const float*)d_in[1];
  const float* vel   = (const float*)d_in[2];
  const float* eattr = (const float*)d_in[3];
  const float* we_w1 = (const float*)d_in[4];
  const float* we_b1 = (const float*)d_in[5];
  const float* we_w2 = (const float*)d_in[6];
  const float* we_b2 = (const float*)d_in[7];
  const float* wx_w1 = (const float*)d_in[8];
  const float* wx_b1 = (const float*)d_in[9];
  const float* wx_w2 = (const float*)d_in[10];
  const float* wx_b2 = (const float*)d_in[11];
  const float* wh_w1 = (const float*)d_in[12];
  const float* wh_b1 = (const float*)d_in[13];
  const float* wh_w2 = (const float*)d_in[14];
  const float* wh_b2 = (const float*)d_in[15];
  const float* wv_w1 = (const float*)d_in[16];
  const float* wv_b1 = (const float*)d_in[17];
  const float* wv_w2 = (const float*)d_in[18];
  const float* wv_b2 = (const float*)d_in[19];
  const int*   arestas = (const int*)d_in[20];

  const int N = in_sizes[0] / 64;
  const int E = in_sizes[20] / 2;

  float* ws     = (float*)d_ws;
  float* m_i    = ws;                      // N*64
  float* aggsum = ws + (size_t)N * 64;     // N*3
  float* cntb   = ws + (size_t)N * 67;     // N

  // zero accumulators
  const long zn = (long)N * 68;
  zero_kernel<<<(int)((zn + 255) / 256), 256, 0, stream>>>(ws, zn);

  // LDS byte counts must match the kernel carving above
  const size_t edge_lds =
      (10240 + 4096 + 4096) * sizeof(f16) + 256 * sizeof(float) +
      (size_t)WV * (2560 + 1024) * sizeof(f16) + (size_t)WV * 1024 * sizeof(float) +
      (size_t)WV * 16 * sizeof(int);
  const size_t node_lds =
      (8192 + 4096 + 4096) * sizeof(f16) + 256 * sizeof(float) +
      (size_t)WV * (2048 + 1024) * sizeof(f16) + (size_t)WV * 1024 * sizeof(float);

  const int edgeBlocks = (E + WV * TPW * 16 - 1) / (WV * TPW * 16);
  edge_kernel<<<edgeBlocks, 256, edge_lds, stream>>>(
      h, x, eattr, arestas, we_w1, we_b1, we_w2, we_b2, wx_w1, wx_b1, wx_w2,
      wx_b2, m_i, aggsum, cntb, N, E);

  const int nodeBlocks = (N + WV * TPW * 16 - 1) / (WV * TPW * 16);
  node_kernel<<<nodeBlocks, 256, node_lds, stream>>>(
      h, x, vel, wh_w1, wh_b1, wh_w2, wh_b2, wv_w1, wv_b1, wv_w2, wv_b2, m_i,
      aggsum, cntb, (float*)d_out, N);
}
